// SA_33294586478784
// MI455X (gfx1250) — matrix operand order repaired and checked by reading, not compiled, not run on hardware
//
#include <hip/hip_runtime.h>
#include <stdint.h>
#include <stddef.h>

// ---------------------------------------------------------------------------
// Types for CDNA5 WMMA (wave32): bf16 16x16x32, f32 accumulate
// ---------------------------------------------------------------------------
typedef __bf16 bf16;
typedef __bf16 bf16x8 __attribute__((ext_vector_type(8)));
typedef __bf16 v16bf  __attribute__((ext_vector_type(16)));
typedef float  v8f    __attribute__((ext_vector_type(8)));

#define SHUF16(lo, hi) __builtin_shufflevector((lo), (hi), 0,1,2,3,4,5,6,7,8,9,10,11,12,13,14,15)

__device__ __forceinline__ v8f wmma_bf16(v16bf a, v16bf b, v8f c) {
    return __builtin_amdgcn_wmma_f32_16x16x32_bf16(false, a, false, b, (short)0, c, false, false);
}

// ---------------------------------------------------------------------------
// fp32 -> bf16 cast (grid-stride)
// ---------------------------------------------------------------------------
__global__ __launch_bounds__(256)
void cast_f32_bf16(const float* __restrict__ in, bf16* __restrict__ out, int n) {
    for (int i = blockIdx.x * blockDim.x + threadIdx.x; i < n; i += gridDim.x * blockDim.x)
        out[i] = (bf16)in[i];
}

// ---------------------------------------------------------------------------
// Transpose + cast: in[K][N] f32  ->  out[N][K] bf16   (K,N multiples of 32)
// ---------------------------------------------------------------------------
__global__ __launch_bounds__(256)
void transpose_cast(const float* __restrict__ in, bf16* __restrict__ out, int K, int N) {
    __shared__ float tile[32][33];
    const int bn = blockIdx.x * 32;           // N
    const int bk = blockIdx.y * 32;           // K
    const int tx = threadIdx.x & 31;
    const int ty = threadIdx.x >> 5;          // 0..7
#pragma unroll
    for (int i = 0; i < 4; ++i) {
        int r = ty + i * 8;
        tile[r][tx] = in[(size_t)(bk + r) * N + bn + tx];
    }
    __syncthreads();
#pragma unroll
    for (int i = 0; i < 4; ++i) {
        int r = ty + i * 8;
        out[(size_t)(bn + r) * K + bk + tx] = (bf16)tile[tx][r];
    }
}

// ---------------------------------------------------------------------------
// Tiled WMMA GEMM: out[M,N] = A[M,K](bf16) * BT[N,K](bf16)^T + bias
// 128x128x32 tiles, 256 threads = 8 waves, each wave owns a 16x128 strip.
// OUT_BF16: store bf16 (activation feeding next GEMM) else f32.
// ---------------------------------------------------------------------------
template <int OUT_BF16, int RELU>
__global__ __launch_bounds__(256)
void gemm_wmma(const bf16* __restrict__ A, const bf16* __restrict__ BT,
               const float* __restrict__ bias, void* __restrict__ out,
               int M, int N, int K) {
    constexpr int BM = 128, BN = 128, BK = 32;
    constexpr int LDA = BK + 8;                  // 40 elems, 80B rows (16B aligned)
    __shared__ __align__(16) bf16 As[2][BM * LDA];
    __shared__ __align__(16) bf16 Bs[2][BN * LDA];

    const int tid   = threadIdx.x;
    const int wave  = tid >> 5;
    const int lane  = tid & 31;
    const int lhalf = lane >> 4;                 // K-half selector per ISA layout
    const int l15   = lane & 15;

    const int bm = blockIdx.y * BM;
    const int bn = blockIdx.x * BN;

    v8f acc[8] = {};

    const int nsteps = K / BK;

    auto stage = [&](int step, int buf) {
        const int k0 = step * BK;
#pragma unroll
        for (int i = 0; i < 2; ++i) {
            int linear = tid + i * 256;          // 0..511
            int row    = linear >> 2;            // 0..127
            int chunk  = (linear & 3) * 8;       // 0,8,16,24
            *(bf16x8*)&As[buf][row * LDA + chunk] =
                *(const bf16x8*)(A + (size_t)(bm + row) * K + k0 + chunk);
            *(bf16x8*)&Bs[buf][row * LDA + chunk] =
                *(const bf16x8*)(BT + (size_t)(bn + row) * K + k0 + chunk);
        }
    };

    stage(0, 0);
    __syncthreads();

    for (int s = 0; s < nsteps; ++s) {
        const int cur = s & 1;
        if (s + 1 < nsteps) stage(s + 1, cur ^ 1);   // prefetch into other buffer

        // A fragment (16x32): lane row = l15, elems 0..7 -> K = 8*lhalf.., 8..15 -> K = 16+8*lhalf..
        const bf16* ar = &As[cur][(wave * 16 + l15) * LDA];
        bf16x8 alo = *(const bf16x8*)(ar + lhalf * 8);
        bf16x8 ahi = *(const bf16x8*)(ar + 16 + lhalf * 8);
        v16bf afrag = SHUF16(alo, ahi);

#pragma unroll
        for (int t = 0; t < 8; ++t) {
            // B fragment (32x16): lane col n = l15, elems 0..7 -> K = 8*lhalf.., 8..15 -> K = 16+8*lhalf..
            const bf16* br = &Bs[cur][(t * 16 + l15) * LDA];
            bf16x8 blo = *(const bf16x8*)(br + lhalf * 8);
            bf16x8 bhi = *(const bf16x8*)(br + 16 + lhalf * 8);
            v16bf bfrag = SHUF16(blo, bhi);
            acc[t] = wmma_bf16(afrag, bfrag, acc[t]);
        }
        __syncthreads();
    }

    // Epilogue: C layout m = r + 8*(lane>=16), n = l15 (+16*t)
    const int mrow0 = bm + wave * 16 + lhalf * 8;
#pragma unroll
    for (int t = 0; t < 8; ++t) {
        const int col = bn + t * 16 + l15;
        const float bv = bias[col];
#pragma unroll
        for (int r = 0; r < 8; ++r) {
            float v = acc[t][r] + bv;
            if (RELU) v = v > 0.0f ? v : 0.0f;
            const size_t off = (size_t)(mrow0 + r) * N + col;
            if (OUT_BF16) ((bf16*)out)[off] = (bf16)v;
            else          ((float*)out)[off] = v;
        }
    }
}

// ---------------------------------------------------------------------------
// Flash attention, bf16 WMMA, fp32 online softmax.
// Block = one (b,h) x 128 query rows, 8 waves (16 q-rows each).
// Q/K/V stored [token, D] bf16 with head columns h*64..h*64+63.
// ---------------------------------------------------------------------------
__global__ __launch_bounds__(256)
void flash_attn(const bf16* __restrict__ Q, const bf16* __restrict__ Kb,
                const bf16* __restrict__ Vb, const uint8_t* __restrict__ mask,
                bf16* __restrict__ O) {
    constexpr int S = 2048, D = 1024, DH = 64;
    constexpr int LKS = DH + 8;      // K tile row stride (elems)
    constexpr int LPS = 128 + 8;     // P tile row stride
    constexpr int LVS = 128 + 8;     // V^T tile row stride

    // smemU is a union: first used as Ks[128][LKS] (9216 elems),
    // then (after a barrier) as per-wave P tiles (8*16*LPS = 17408 elems).
    __shared__ __align__(16) bf16 smemU[8 * 16 * LPS];
    __shared__ __align__(16) bf16 Vts[64 * LVS];

    const int qb0 = blockIdx.x * 128;
    const int bh  = blockIdx.y;
    const int b   = bh >> 4;
    const int h   = bh & 15;
    const size_t base = (size_t)b * S * D + (size_t)h * DH;
    const uint8_t* mrow = mask + (size_t)b * S;

    const int tid = threadIdx.x;
    const int wave = tid >> 5, lane = tid & 31;
    const int lhalf = lane >> 4, l15 = lane & 15;

    bf16 (*Ks)[LKS] = (bf16(*)[LKS])smemU;
    bf16* Pw = smemU + wave * 16 * LPS;

    // Preload Q fragments for this wave's 16 rows (d = 0..63 -> 2 WMMA K-steps)
    v16bf aq[2];
    {
        const bf16* qrow = Q + base + (size_t)(qb0 + wave * 16 + l15) * D;
#pragma unroll
        for (int kk = 0; kk < 2; ++kk) {
            bf16x8 lo = *(const bf16x8*)(qrow + kk * 32 + lhalf * 8);
            bf16x8 hi = *(const bf16x8*)(qrow + kk * 32 + 16 + lhalf * 8);
            aq[kk] = SHUF16(lo, hi);
        }
    }

    float m_i[8], l_i[8];
#pragma unroll
    for (int r = 0; r < 8; ++r) { m_i[r] = -1e30f; l_i[r] = 0.0f; }
    v8f cacc[4] = {};

    for (int j = 0; j < S / 128; ++j) {
        __syncthreads();   // previous iteration done reading Ps/Vts

        // Stage K tile [128][64] and V^T tile [64][128]
#pragma unroll
        for (int i = 0; i < 4; ++i) {
            int linear = tid + i * 256;            // 0..1023
            int row    = linear >> 3;              // 0..127
            int chunk  = (linear & 7) * 8;         // 0..56
            *(bf16x8*)&Ks[row][chunk] =
                *(const bf16x8*)(Kb + base + (size_t)(j * 128 + row) * D + chunk);
            bf16x8 vv =
                *(const bf16x8*)(Vb + base + (size_t)(j * 128 + row) * D + chunk);
#pragma unroll
            for (int e = 0; e < 8; ++e) Vts[(chunk + e) * LVS + row] = vv[e];
        }
        __syncthreads();

        // scores = Q * K^T  (B fragment rows are K rows: n = kpos, d in same order as A)
        v8f sacc[8] = {};
#pragma unroll
        for (int kk = 0; kk < 2; ++kk) {
#pragma unroll
            for (int t = 0; t < 8; ++t) {
                const bf16* br = &Ks[t * 16 + l15][kk * 32];
                bf16x8 blo = *(const bf16x8*)(br + lhalf * 8);
                bf16x8 bhi = *(const bf16x8*)(br + 16 + lhalf * 8);
                sacc[t] = wmma_bf16(aq[kk], SHUF16(blo, bhi), sacc[t]);
            }
        }

        // scale 1/sqrt(DH) + mask
#pragma unroll
        for (int t = 0; t < 8; ++t) {
            const float madd = mrow[j * 128 + t * 16 + l15] ? -1e9f : 0.0f;
#pragma unroll
            for (int r = 0; r < 8; ++r) sacc[t][r] = sacc[t][r] * 0.125f + madd;
        }

        // online softmax per row (rows live in 16-lane halves; xor<=8 stays in half)
        float corr[8];
#pragma unroll
        for (int r = 0; r < 8; ++r) {
            float rm = -1e30f;
#pragma unroll
            for (int t = 0; t < 8; ++t) rm = fmaxf(rm, sacc[t][r]);
            for (int msk = 8; msk >= 1; msk >>= 1) rm = fmaxf(rm, __shfl_xor(rm, msk, 32));
            const float mnew = fmaxf(m_i[r], rm);
            const float c = __expf(m_i[r] - mnew);
            m_i[r] = mnew;
            float rs = 0.0f;
#pragma unroll
            for (int t = 0; t < 8; ++t) {
                const float p = __expf(sacc[t][r] - mnew);
                sacc[t][r] = p;
                rs += p;
            }
            for (int msk = 8; msk >= 1; msk >>= 1) rs += __shfl_xor(rs, msk, 32);
            l_i[r] = l_i[r] * c + rs;
            corr[r] = c;
        }
#pragma unroll
        for (int t = 0; t < 4; ++t)
#pragma unroll
            for (int r = 0; r < 8; ++r) cacc[t][r] *= corr[r];

        __syncthreads();   // everyone done reading Ks before P overwrites the union

        // Spill P (bf16) to this wave's LDS tile in row-major [m][kpos]
#pragma unroll
        for (int t = 0; t < 8; ++t)
#pragma unroll
            for (int r = 0; r < 8; ++r)
                Pw[(r + 8 * lhalf) * LPS + t * 16 + l15] = (bf16)sacc[t][r];

        // ctx += P(16x128) * V(128x64): 4 K-steps over kpos, 4 d-tiles
#pragma unroll
        for (int kk = 0; kk < 4; ++kk) {
            const bf16* pr = Pw + l15 * LPS + kk * 32;
            bf16x8 plo = *(const bf16x8*)(pr + lhalf * 8);
            bf16x8 phi = *(const bf16x8*)(pr + 16 + lhalf * 8);
            v16bf ap = SHUF16(plo, phi);
#pragma unroll
            for (int t = 0; t < 4; ++t) {
                const bf16* br = Vts + (t * 16 + l15) * LVS + kk * 32;
                bf16x8 blo = *(const bf16x8*)(br + lhalf * 8);
                bf16x8 bhi = *(const bf16x8*)(br + 16 + lhalf * 8);
                cacc[t] = wmma_bf16(ap, SHUF16(blo, bhi), cacc[t]);
            }
        }
    }

    // normalize and store ctx (bf16) back to [token, D] at head columns
#pragma unroll
    for (int r = 0; r < 8; ++r) {
        const float inv = 1.0f / l_i[r];
        const int token = qb0 + wave * 16 + r + 8 * lhalf;
        bf16* orow = O + base + (size_t)token * D;
#pragma unroll
        for (int t = 0; t < 4; ++t)
            orow[t * 16 + l15] = (bf16)(cacc[t][r] * inv);
    }
}

// ---------------------------------------------------------------------------
// Fused residual + LayerNorm over D=1024 (one row per block, 256 threads)
// outf = LN(x + y); optionally also emit bf16 copy for the next GEMM.
// ---------------------------------------------------------------------------
template <int WRITE_BF16>
__global__ __launch_bounds__(256)
void residual_ln(const float* __restrict__ x, const float* __restrict__ y,
                 const float* __restrict__ g, const float* __restrict__ b,
                 float* __restrict__ outf, bf16* __restrict__ outb) {
    constexpr int D = 1024;
    const int row = blockIdx.x;
    const int tid = threadIdx.x;
    __shared__ float red[8];

    float v[4];
    float s = 0.0f;
#pragma unroll
    for (int i = 0; i < 4; ++i) {
        const int c = tid + i * 256;
        v[i] = x[(size_t)row * D + c] + y[(size_t)row * D + c];
        s += v[i];
    }
    for (int m = 16; m >= 1; m >>= 1) s += __shfl_xor(s, m, 32);
    if ((tid & 31) == 0) red[tid >> 5] = s;
    __syncthreads();
    float tot = 0.0f;
#pragma unroll
    for (int w = 0; w < 8; ++w) tot += red[w];
    const float mu = tot * (1.0f / (float)D);
    __syncthreads();

    float vs = 0.0f;
#pragma unroll
    for (int i = 0; i < 4; ++i) {
        const float d = v[i] - mu;
        vs += d * d;
    }
    for (int m = 16; m >= 1; m >>= 1) vs += __shfl_xor(vs, m, 32);
    if ((tid & 31) == 0) red[tid >> 5] = vs;
    __syncthreads();
    float tot2 = 0.0f;
#pragma unroll
    for (int w = 0; w < 8; ++w) tot2 += red[w];
    const float rstd = rsqrtf(tot2 * (1.0f / (float)D) + 1e-6f);

#pragma unroll
    for (int i = 0; i < 4; ++i) {
        const int c = tid + i * 256;
        const float o = (v[i] - mu) * rstd * g[c] + b[c];
        outf[(size_t)row * D + c] = o;
        if (WRITE_BF16) outb[(size_t)row * D + c] = (bf16)o;
    }
}

// ---------------------------------------------------------------------------
// Launch: full encoder layer
// ---------------------------------------------------------------------------
extern "C" void kernel_launch(void* const* d_in, const int* in_sizes, int n_in,
                              void* d_out, int out_size, void* d_ws, size_t ws_size,
                              hipStream_t stream) {
    (void)in_sizes; (void)n_in; (void)out_size; (void)ws_size;

    const float*   x    = (const float*)d_in[0];
    const uint8_t* mask = (const uint8_t*)d_in[1];
    const float* wq = (const float*)d_in[2];  const float* bq = (const float*)d_in[3];
    const float* wk = (const float*)d_in[4];  const float* bk = (const float*)d_in[5];
    const float* wv = (const float*)d_in[6];  const float* bv = (const float*)d_in[7];
    const float* wm = (const float*)d_in[8];  const float* bm = (const float*)d_in[9];
    const float* w1 = (const float*)d_in[10]; const float* b1 = (const float*)d_in[11];
    const float* w2 = (const float*)d_in[12]; const float* b2 = (const float*)d_in[13];
    const float* g1 = (const float*)d_in[14]; const float* be1 = (const float*)d_in[15];
    const float* g2 = (const float*)d_in[16]; const float* be2 = (const float*)d_in[17];
    float* out = (float*)d_out;

    constexpr int NT = 8192, D = 1024, FF = 4096, S = 2048;
    char* ws = (char*)d_ws;
    const size_t MB = 1024 * 1024;
    bf16*  xb   = (bf16*) (ws + 0);        // 16 MB
    bf16*  wqT  = (bf16*) (ws + 16 * MB);  //  2 MB
    bf16*  wkT  = (bf16*) (ws + 18 * MB);
    bf16*  wvT  = (bf16*) (ws + 20 * MB);
    bf16*  wmT  = (bf16*) (ws + 22 * MB);
    bf16*  w1T  = (bf16*) (ws + 24 * MB);  //  8 MB
    bf16*  w2T  = (bf16*) (ws + 32 * MB);  //  8 MB
    bf16*  qb   = (bf16*) (ws + 40 * MB);  // 16 MB
    bf16*  kb   = (bf16*) (ws + 56 * MB);
    bf16*  vb   = (bf16*) (ws + 72 * MB);
    bf16*  ctxb = (bf16*) (ws + 88 * MB);
    float* attf = (float*)(ws + 104 * MB); // 32 MB (reused for FFN output)
    float* x1f  = (float*)(ws + 136 * MB); // 32 MB
    bf16*  x1b  = (bf16*) (ws + 168 * MB); // 16 MB
    bf16*  hb   = (bf16*) (ws + 184 * MB); // 64 MB  -> total 248 MB

    // bf16 casts / weight pre-transposes (cheap vs GEMMs)
    cast_f32_bf16<<<4096, 256, 0, stream>>>(x, xb, NT * D);
    transpose_cast<<<dim3(D / 32,  D / 32),  256, 0, stream>>>(wq, wqT, D, D);
    transpose_cast<<<dim3(D / 32,  D / 32),  256, 0, stream>>>(wk, wkT, D, D);
    transpose_cast<<<dim3(D / 32,  D / 32),  256, 0, stream>>>(wv, wvT, D, D);
    transpose_cast<<<dim3(D / 32,  D / 32),  256, 0, stream>>>(wm, wmT, D, D);
    transpose_cast<<<dim3(FF / 32, D / 32),  256, 0, stream>>>(w1, w1T, D, FF);
    transpose_cast<<<dim3(D / 32,  FF / 32), 256, 0, stream>>>(w2, w2T, FF, D);

    // QKV projections
    gemm_wmma<1, 0><<<dim3(D / 128, NT / 128), 256, 0, stream>>>(xb, wqT, bq, qb, NT, D, D);
    gemm_wmma<1, 0><<<dim3(D / 128, NT / 128), 256, 0, stream>>>(xb, wkT, bk, kb, NT, D, D);
    gemm_wmma<1, 0><<<dim3(D / 128, NT / 128), 256, 0, stream>>>(xb, wvT, bv, vb, NT, D, D);

    // attention
    flash_attn<<<dim3(S / 128, 4 * 16), 256, 0, stream>>>(qb, kb, vb, mask, ctxb);

    // out-proj + residual + LN1
    gemm_wmma<0, 0><<<dim3(D / 128, NT / 128), 256, 0, stream>>>(ctxb, wmT, bm, attf, NT, D, D);
    residual_ln<1><<<NT, 256, 0, stream>>>(x, attf, g1, be1, x1f, x1b);

    // FFN + residual + LN2
    gemm_wmma<1, 1><<<dim3(FF / 128, NT / 128), 256, 0, stream>>>(x1b, w1T, b1, hb, NT, FF, D);
    gemm_wmma<0, 0><<<dim3(D / 128, NT / 128), 256, 0, stream>>>(hb, w2T, b2, attf, NT, D, FF);
    residual_ln<0><<<NT, 256, 0, stream>>>(x1f, attf, g2, be2, out, nullptr);
}
